// HOI_59047210385536
// MI455X (gfx1250) — matrix operand order repaired and checked by reading, not compiled, not run on hardware
//
#include <hip/hip_runtime.h>
#include <hip/hip_bf16.h>

// ---------------------------------------------------------------------------
// Soft vector quantization for MI455X (gfx1250), wave32 + WMMA bf16.
//   feat: (16,512,32,32) f32, codebook: (4096,512) f32
//   outputs (concat): q_feat (16,512,32,32) | assignment (16,4096,32,32) |
//                     distance (16384,4096)   -- all f32
// ---------------------------------------------------------------------------

typedef __bf16 bf16;
typedef __attribute__((ext_vector_type(16))) __bf16 v16bf;
typedef __attribute__((ext_vector_type(8)))  __bf16 v8bf;
typedef __attribute__((ext_vector_type(8)))  float  v8f;

#define N_ROWS   16384   // b*h*w
#define C_DIM    512
#define K_DIM    4096
#define HW       1024    // h*w per batch
#define PS       4104    // padded LDS stride for p (K + 8) -> no bank conflicts
#define INVT     (1.0f / 0.07f)

#define WMMA_BF16(a, b, c) \
    __builtin_amdgcn_wmma_f32_16x16x32_bf16(false, (a), false, (b), (short)0, (c), false, false)

#define CAT16(lo, hi) __builtin_shufflevector((lo), (hi), 0,1,2,3,4,5,6,7,8,9,10,11,12,13,14,15)

// ---------------- prep: codebook f32 -> bf16 (row major, coalesced) --------
__global__ void k_cvt_cb(const float* __restrict__ cbf, bf16* __restrict__ cbh) {
    int gid = blockIdx.x * 256 + threadIdx.x;        // 1024 blocks, 8 elems/thread
    const float4* s = (const float4*)cbf + (size_t)gid * 2;
    float4 f0 = s[0], f1 = s[1];
    v8bf o;
    o[0] = (bf16)f0.x; o[1] = (bf16)f0.y; o[2] = (bf16)f0.z; o[3] = (bf16)f0.w;
    o[4] = (bf16)f1.x; o[5] = (bf16)f1.y; o[6] = (bf16)f1.z; o[7] = (bf16)f1.w;
    ((v8bf*)cbh)[gid] = o;
}

// ---------------- prep: codebook f32 (K,C) -> bf16 transposed (C,K) --------
__global__ void k_tr_cb(const float* __restrict__ cbf, bf16* __restrict__ cbt) {
    __shared__ bf16 tile[32][33];
    int kt = blockIdx.x * 32, ct = blockIdx.y * 32;
    int x = threadIdx.x & 31, y0 = threadIdx.x >> 5;
    for (int yy = y0; yy < 32; yy += 8)
        tile[yy][x] = (bf16)cbf[(size_t)(kt + yy) * C_DIM + ct + x];
    __syncthreads();
    for (int yy = y0; yy < 32; yy += 8)
        cbt[(size_t)(ct + yy) * K_DIM + kt + x] = tile[x][yy];
}

// ---------------- prep: feat (b,c,hw) f32 -> xb (b*hw, c) bf16 -------------
__global__ void k_tr_feat(const float* __restrict__ feat, bf16* __restrict__ xb) {
    __shared__ bf16 tile[32][33];
    int b  = blockIdx.z;
    int ht = blockIdx.x * 32;          // hw tile
    int ct = blockIdx.y * 32;          // channel tile
    int x = threadIdx.x & 31, y0 = threadIdx.x >> 5;
    const float* src = feat + (size_t)b * C_DIM * HW;
    for (int yy = y0; yy < 32; yy += 8)
        tile[yy][x] = (bf16)src[(size_t)(ct + yy) * HW + ht + x];
    __syncthreads();
    for (int yy = y0; yy < 32; yy += 8)
        xb[((size_t)b * HW + ht + yy) * C_DIM + ct + x] = tile[x][yy];
}

// ---------------- prep: row norms (f32, from original f32 data) ------------
__global__ void k_norms_x(const float* __restrict__ feat, float* __restrict__ xn) {
    int n = blockIdx.x * 256 + threadIdx.x;          // 64 blocks
    int b = n >> 10, hw = n & 1023;
    const float* p = feat + (size_t)b * C_DIM * HW + hw;
    float s = 0.f;
    for (int c = 0; c < C_DIM; ++c) { float v = p[(size_t)c * HW]; s += v * v; }
    xn[n] = s;
}

__global__ void k_norms_e(const float* __restrict__ cbf, float* __restrict__ en) {
    int k = blockIdx.x * 256 + threadIdx.x;          // 16 blocks
    const float* p = cbf + (size_t)k * C_DIM;
    float s = 0.f;
    for (int c = 0; c < C_DIM; ++c) { float v = p[c]; s += v * v; }
    en[k] = s;
}

// ---------------------------------------------------------------------------
// main fused kernel: one workgroup = 16 token rows x full K.
//   phase 1: distance tiles via v_wmma_f32_16x16x32_bf16 (2 N-tiles per pass,
//            2 independent accumulator chains), coalesced staged store
//   phase 2: row min + exp sums (L2-warm re-read of own distance rows)
//   phase 3: assignment softmax (coalesced over hw) + p (bf16) into LDS
//   phase 4: q = p @ codebook via WMMA (4 C-tiles per K-sweep, 4 chains)
// ---------------------------------------------------------------------------
__launch_bounds__(256)
__global__ void vq_main(const bf16* __restrict__ xb, const float* __restrict__ xn,
                        const bf16* __restrict__ cb, const bf16* __restrict__ cbt,
                        const float* __restrict__ en,
                        float* __restrict__ out_q, float* __restrict__ out_assign,
                        float* __restrict__ out_dist) {
    extern __shared__ char smem[];
    bf16*  xs    = (bf16*)(smem);                 // 16 x 512 bf16     : 16384 B
    bf16*  ps    = (bf16*)(smem + 16384);         // 16 x PS bf16      : 131328 B
    float* stage = (float*)(smem + 147712);       // 8 waves x 256 f32 : 8192 B
    float* red0  = (float*)(smem + 155904);       // 16 x 16 f32
    float* red1  = (float*)(smem + 156928);       // 16 x 16 f32
    float* rmin  = (float*)(smem + 157952);       // 16 f32
    float* rinv  = (float*)(smem + 158016);       // 16 f32 (1/sum1), 32B aligned
    float* rtin  = (float*)(smem + 158080);       // 16 f32 (1/sumT)

    const int t    = threadIdx.x;
    const int wave = t >> 5;
    const int lane = t & 31;
    const int half = lane >> 4;
    const int nl   = lane & 15;

    const int m0   = blockIdx.x * 16;     // first token row
    const int bidx = m0 >> 10;            // batch
    const int hw0  = m0 & 1023;           // h*w offset inside batch

    // ---- load x tile (16 x 512 bf16 = 16 KB) into LDS, coalesced ----------
    {
        const uint4* src = (const uint4*)(xb + (size_t)m0 * C_DIM);
        uint4* dst = (uint4*)xs;
        for (int i = t; i < 1024; i += 256) dst[i] = src[i];
    }
    __syncthreads();

    // ================= phase 1: distance = ||x||^2+||e||^2-2 x.e ==========
    // wave w owns codes [w*512, w*512+512): 16 passes x 2 tiles of 16 codes
    {
        v8f xnv = *(const v8f*)(xn + m0 + half * 8);   // norms for my 8 rows
        float* st = stage + wave * 256;                // per-wave 16x16 staging
        for (int it = 0; it < 16; ++it) {
            const int n0     = wave * 512 + it * 32;
            const int ncode0 = n0 + nl;
            const int ncode1 = n0 + 16 + nl;
            const bf16* bb0 = cb + (size_t)ncode0 * C_DIM + half * 16;
            const bf16* bb1 = cb + (size_t)ncode1 * C_DIM + half * 16;
            v8f acc0 = {}, acc1 = {};
#pragma unroll
            for (int cc = 0; cc < 16; ++cc) {
                const int c0 = cc * 32;
                const bf16* ap = xs + nl * C_DIM + c0 + half * 16;
                v8bf alo = *(const v8bf*)(ap);
                v8bf ahi = *(const v8bf*)(ap + 8);
                v16bf a  = CAT16(alo, ahi);
                v16bf b0 = *(const v16bf*)(bb0 + c0);
                v16bf b1 = *(const v16bf*)(bb1 + c0);
                acc0 = WMMA_BF16(a, b0, acc0);      // two independent chains
                acc1 = WMMA_BF16(a, b1, acc1);
            }
            const float e0 = en[ncode0];
            const float e1 = en[ncode1];
            // ---- tile 0: stage (transpose in LDS) then coalesced store ----
#pragma unroll
            for (int v = 0; v < 8; ++v) {
                const int m = half * 8 + v;
                st[m * 16 + nl] = xnv[v] + e0 - 2.0f * acc0[v];
            }
            asm volatile("s_wait_dscnt 0" ::: "memory");
            __builtin_amdgcn_wave_barrier();
#pragma unroll
            for (int i = 0; i < 8; ++i) {
                const int row = i * 2 + half;   // two 64B rows per store
                out_dist[(size_t)(m0 + row) * K_DIM + n0 + nl] = st[row * 16 + nl];
            }
            asm volatile("s_wait_dscnt 0" ::: "memory");   // reads done before
            __builtin_amdgcn_wave_barrier();               // tile-1 overwrites
            // ---- tile 1 ---------------------------------------------------
#pragma unroll
            for (int v = 0; v < 8; ++v) {
                const int m = half * 8 + v;
                st[m * 16 + nl] = xnv[v] + e1 - 2.0f * acc1[v];
            }
            asm volatile("s_wait_dscnt 0" ::: "memory");
            __builtin_amdgcn_wave_barrier();
#pragma unroll
            for (int i = 0; i < 8; ++i) {
                const int row = i * 2 + half;
                out_dist[(size_t)(m0 + row) * K_DIM + n0 + 16 + nl] = st[row * 16 + nl];
            }
            asm volatile("s_wait_dscnt 0" ::: "memory");
            __builtin_amdgcn_wave_barrier();
        }
    }
    __threadfence_block();     // make this WG's distance rows visible to itself
    __syncthreads();

    // ================= phase 2: row min + exp sums (L2-warm) ==============
    {
        const int g = t & 15;                  // worker within row
        const int m = t >> 4;                  // row
        const float* dr = out_dist + (size_t)(m0 + m) * K_DIM;
        float pm = 3.0e38f;
        for (int i = 0; i < 256; ++i) pm = fminf(pm, dr[g + 16 * i]);
        red0[m * 16 + g] = pm;
        __syncthreads();
        if (t < 16) {
            float mn = red0[t * 16];
            for (int j = 1; j < 16; ++j) mn = fminf(mn, red0[t * 16 + j]);
            rmin[t] = mn;
        }
        __syncthreads();
        const float dmin = rmin[m];
        float s1 = 0.f, sT = 0.f;
        for (int i = 0; i < 256; ++i) {
            const float z = dmin - dr[g + 16 * i];
            s1 += __expf(z);
            sT += __expf(z * INVT);
        }
        red0[m * 16 + g] = s1;
        red1[m * 16 + g] = sT;
        __syncthreads();
        if (t < 16) {
            float a1 = 0.f, aT = 0.f;
            for (int j = 0; j < 16; ++j) { a1 += red0[t * 16 + j]; aT += red1[t * 16 + j]; }
            rinv[t] = 1.0f / a1;
            rtin[t] = 1.0f / aT;
        }
        __syncthreads();
    }

    // ====== phase 3: assignment (coalesced over hw) + p(bf16) into LDS =====
    {
        const int mc = t & 15;                 // row -> lanes give consecutive hw
        const int kg = t >> 4;
        const float dmin = rmin[mc];
        const float sti  = rtin[mc];
        const float* dr  = out_dist + (size_t)(m0 + mc) * K_DIM;
        float* ab = out_assign + (size_t)bidx * K_DIM * HW + hw0 + mc;
        for (int i = 0; i < 256; ++i) {
            const int k = kg + 16 * i;
            const float z = dmin - dr[k];
            ps[mc * PS + k] = (bf16)__expf(z);          // unnormalized p
            ab[(size_t)k * HW] = __expf(z * INVT) * sti;
        }
    }
    __syncthreads();

    // ================= phase 4: q = (p / sum1) @ codebook ==================
    // wave w owns channels [w*64, w*64+64): 4 tiles, one shared K-sweep,
    // 4 independent accumulator chains, A loaded once per k-chunk.
    {
        v8f inv = *(const v8f*)(rinv + half * 8);
        const int c0 = wave * 64;
        const bf16* bb0 = cbt + (size_t)(c0      + nl) * K_DIM + half * 16;
        const bf16* bb1 = cbt + (size_t)(c0 + 16 + nl) * K_DIM + half * 16;
        const bf16* bb2 = cbt + (size_t)(c0 + 32 + nl) * K_DIM + half * 16;
        const bf16* bb3 = cbt + (size_t)(c0 + 48 + nl) * K_DIM + half * 16;
        v8f acc0 = {}, acc1 = {}, acc2 = {}, acc3 = {};
#pragma unroll 2
        for (int kk = 0; kk < 128; ++kk) {
            const int k0 = kk * 32;
            const bf16* ap = ps + nl * PS + k0 + half * 16;
            v8bf alo = *(const v8bf*)(ap);
            v8bf ahi = *(const v8bf*)(ap + 8);
            v16bf a  = CAT16(alo, ahi);
            v16bf b0 = *(const v16bf*)(bb0 + k0);
            v16bf b1 = *(const v16bf*)(bb1 + k0);
            v16bf b2 = *(const v16bf*)(bb2 + k0);
            v16bf b3 = *(const v16bf*)(bb3 + k0);
            acc0 = WMMA_BF16(a, b0, acc0);
            acc1 = WMMA_BF16(a, b1, acc1);
            acc2 = WMMA_BF16(a, b2, acc2);
            acc3 = WMMA_BF16(a, b3, acc3);
        }
        float* qbase = out_q + (size_t)bidx * C_DIM * HW + hw0 + half * 8;
        v8f dq;
#pragma unroll
        for (int v = 0; v < 8; ++v) dq[v] = acc0[v] * inv[v];
        *(v8f*)(qbase + (size_t)(c0      + nl) * HW) = dq;
#pragma unroll
        for (int v = 0; v < 8; ++v) dq[v] = acc1[v] * inv[v];
        *(v8f*)(qbase + (size_t)(c0 + 16 + nl) * HW) = dq;
#pragma unroll
        for (int v = 0; v < 8; ++v) dq[v] = acc2[v] * inv[v];
        *(v8f*)(qbase + (size_t)(c0 + 32 + nl) * HW) = dq;
#pragma unroll
        for (int v = 0; v < 8; ++v) dq[v] = acc3[v] * inv[v];
        *(v8f*)(qbase + (size_t)(c0 + 48 + nl) * HW) = dq;
    }
}

// ---------------------------------------------------------------------------
extern "C" void kernel_launch(void* const* d_in, const int* in_sizes, int n_in,
                              void* d_out, int out_size, void* d_ws, size_t ws_size,
                              hipStream_t stream) {
    (void)in_sizes; (void)n_in; (void)out_size; (void)ws_size;

    const float* feat     = (const float*)d_in[0];   // (16,512,32,32)
    const float* codebook = (const float*)d_in[1];   // (4096,512)

    // workspace carve-up (~24.1 MB)
    char* ws = (char*)d_ws;
    bf16*  xb  = (bf16*)(ws);                        // 16384 x 512 bf16 : 16 MB
    bf16*  cbh = (bf16*)(ws + 16777216);             // 4096 x 512 bf16  : 4 MB
    bf16*  cbt = (bf16*)(ws + 20971520);             // 512 x 4096 bf16  : 4 MB
    float* xn  = (float*)(ws + 25165824);            // 16384 f32
    float* en  = (float*)(ws + 25231360);            // 4096 f32

    float* out_q      = (float*)d_out;               // 8388608 f32
    float* out_assign = out_q + 8388608;             // 67108864 f32
    float* out_dist   = out_assign + 67108864;       // 67108864 f32

    // prep
    k_cvt_cb  <<<1024, 256, 0, stream>>>(codebook, cbh);
    k_tr_cb   <<<dim3(128, 16), 256, 0, stream>>>(codebook, cbt);
    k_tr_feat <<<dim3(32, 16, 16), 256, 0, stream>>>(feat, xb);
    k_norms_x <<<64, 256, 0, stream>>>(feat, xn);
    k_norms_e <<<16, 256, 0, stream>>>(codebook, en);

    // fused distance + double softmax + weighted-sum quantization
    vq_main<<<1024, 256, 158144, stream>>>(xb, xn, cbh, cbt, en,
                                           out_q, out_assign, out_dist);
}
